// apexTransducerLoss_38010460569796
// MI455X (gfx1250) — compile-verified, matrix operand order checked by reading, not run on hardware
//
#include <hip/hip_runtime.h>
#include <hip/hip_bf16.h>

#define Bsz 8
#define Tsz 512
#define Usz 128
#define U1  129
#define Vsz 128
#define TOTAL_ROWS (Bsz * Tsz * U1)   /* 528384, divisible by 64 */

#define NEG_INF (-__builtin_inff())

typedef _Float16 v16h __attribute__((ext_vector_type(16)));
typedef float    v8f  __attribute__((ext_vector_type(8)));

// ---------------------------------------------------------------------------
// Kernel 1: fused log-softmax stats + blank/label gather.
// 4 waves / block; each wave owns 16 rows of 128 fp32 logits.
//   stage 1: async global->LDS staging (global_load_async_to_lds_b128,
//            per-lane LDS dest + per-lane global addr, drained via ASYNCcnt)
//   stage 2: per-row max (lane = row%16, half = lane/16; combine via shfl_xor 16)
//   stage 3: exp(x-m) -> f16 packed in the 16x32 A-fragment layout,
//            row-sum via 4 chained v_wmma_f32_16x16x32_f16 vs all-ones B
//   stage 4: lanes 0..15 emit blank = x[0]-lse and emit = x[y[b,u]]-lse
// ---------------------------------------------------------------------------
#define K1_WAVES 4
#define ROW_PAD  132   /* dwords per LDS row: 128 + 4 -> bank stride 4, no conflicts on b128 */

__global__ __launch_bounds__(128) void k1_softmax_gather(
    const float* __restrict__ logits,
    const int*   __restrict__ y,
    float*       __restrict__ blank_ws,
    float*       __restrict__ emit_ws)
{
    __shared__ float lds[K1_WAVES * 16 * ROW_PAD];

    const int lane   = threadIdx.x & 31;
    const int wave   = threadIdx.x >> 5;
    const int tileId = blockIdx.x * K1_WAVES + wave;
    const int rowBase = tileId * 16;
    if (rowBase >= TOTAL_ROWS) return;   // uniform per wave: EXEC stays all-ones

    float* slab = lds + wave * 16 * ROW_PAD;

    // ---- stage 1: async-copy 16 rows x 128 f32 into LDS ----
    // Flat pointers to LDS are {aperture_hi, lds_offset}: low 32 bits are the
    // wave-relative LDS byte address the async engine wants in VDST.
    const unsigned slabLds = (unsigned)(unsigned long long)(void*)slab;
    const float* src = logits + (size_t)rowBase * Vsz;
    #pragma unroll
    for (int r = 0; r < 16; ++r) {
        unsigned long long ga = (unsigned long long)(src + r * Vsz + lane * 4);
        unsigned la = slabLds + (unsigned)(r * (ROW_PAD * 4) + lane * 16);
        asm volatile("global_load_async_to_lds_b128 %0, %1, off"
                     :: "v"(la), "v"(ga) : "memory");
    }
#if __has_builtin(__builtin_amdgcn_s_wait_asynccnt)
    __builtin_amdgcn_s_wait_asynccnt(0);
#else
    asm volatile("s_wait_asynccnt 0" ::: "memory");
#endif

    // ---- stage 2: row max. lane owns row r = lane&15, column-half g = lane>>4
    const int r = lane & 15;
    const int g = lane >> 4;
    const float* myrow = slab + r * ROW_PAD;

    float m = NEG_INF;
    #pragma unroll
    for (int k0 = 0; k0 < 128; k0 += 32) {
        #pragma unroll
        for (int j = 0; j < 8; ++j) {
            m = fmaxf(m, myrow[k0 + g * 8 + j]);
            m = fmaxf(m, myrow[k0 + 16 + g * 8 + j]);
        }
    }
    m = fmaxf(m, __shfl_xor(m, 16, 32));   // full-row max on both column halves

    // ---- stage 3: exp(x-m) -> A fragments, row-sum via WMMA * ones ----
    v16h bones;
    #pragma unroll
    for (int i = 0; i < 16; ++i) bones[i] = (_Float16)1.0f;

    v8f acc = {};
    #pragma unroll
    for (int k0 = 0; k0 < 128; k0 += 32) {
        v16h afrag;
        // 16x32 f16 A layout: lanes 0-15 hold K = k0+{0..7,16..23},
        //                     lanes 16-31 hold K = k0+{8..15,24..31}
        #pragma unroll
        for (int j = 0; j < 8; ++j) {
            afrag[j]     = (_Float16)__expf(myrow[k0 + g * 8 + j]      - m);
            afrag[8 + j] = (_Float16)__expf(myrow[k0 + 16 + g * 8 + j] - m);
        }
        acc = __builtin_amdgcn_wmma_f32_16x16x32_f16(
                  false, afrag, false, bones, (short)0, acc, false, false);
    }

    // C/D layout: acc[i] on lane L holds row M = i + 8*(L>=16), any column.
    float v = acc[r & 7];
    float w = __shfl_xor(v, 16, 32);
    float s = (((r >= 8) ? 1 : 0) == g) ? v : w;   // pick the half holding row r
    float lse = m + __logf(s);                      // log-sum-exp of the row

    // ---- stage 4: write blank / emit (lanes 0..15, one row each) ----
    if (lane < 16) {
        int row = rowBase + lane;          // == (b*T + t)*(U+1) + u
        int bt  = row / U1;
        int u   = row - bt * U1;
        int b   = bt / Tsz;
        const float* rr = slab + lane * ROW_PAD;

        blank_ws[row] = rr[0] - lse;
        if (u < Usz) {
            int lab = y[b * Usz + u];
            emit_ws[bt * Usz + u] = rr[lab] - lse;
        }
    }
}

// ---------------------------------------------------------------------------
// Kernel 2: alpha recurrence as an anti-diagonal wavefront.
// One block per batch element. Cell (t,u) on diagonal d = t+u depends only on
// diagonal d-1, so each diagonal (<=129 cells) is fully parallel; one barrier
// per diagonal, 639 diagonals total. blank/emit come out of L2 (ws ~4 MB).
// ---------------------------------------------------------------------------
__device__ __forceinline__ float logaddexpf_(float x, float y) {
    float mx = fmaxf(x, y);
    if (mx == NEG_INF) return NEG_INF;
    float mn = fminf(x, y);
    return mx + log1pf(__expf(mn - mx));
}

#define K2_THREADS 160

__global__ __launch_bounds__(K2_THREADS) void k2_alpha_wavefront(
    const float* __restrict__ blank_ws,
    const float* __restrict__ emit_ws,
    const int*   __restrict__ logit_lens,
    const int*   __restrict__ y_lens,
    float*       __restrict__ negll)
{
    __shared__ float buf[2][U1];

    const int b = blockIdx.x;
    const int u = threadIdx.x;
    const int tLast = logit_lens[b] - 1;
    const int uLast = y_lens[b];

    const float* blk = blank_ws + (size_t)b * Tsz * U1;
    const float* emt = emit_ws  + (size_t)b * Tsz * Usz;

    if (u <= Usz) buf[0][u] = (u == 0) ? 0.0f : NEG_INF;   // alpha[0,0] = 0
    if (u == 0 && tLast == 0 && uLast == 0)                // degenerate path
        negll[b] = -(0.0f + blk[0]);
    __syncthreads();

    for (int d = 1; d <= (Tsz - 1) + Usz; ++d) {
        const float* prev = buf[(d - 1) & 1];
        float*       cur  = buf[d & 1];

        float val = NEG_INF;
        int t = d - u;
        bool active = (u <= Usz) && (t >= 0) && (t <= Tsz - 1);
        if (active) {
            float below = (t >= 1) ? prev[u]     + blk[(t - 1) * U1 + u]  : NEG_INF;
            float left  = (u >= 1) ? prev[u - 1] + emt[t * Usz + (u - 1)] : NEG_INF;
            val = logaddexpf_(below, left);
            if (t == tLast && u == uLast)
                negll[b] = -(val + blk[tLast * U1 + uLast]);
        }
        if (u <= Usz) cur[u] = val;   // inactive slots become -inf (never read)
        __syncthreads();
    }
}

// ---------------------------------------------------------------------------
// Kernel 3: mean over the 8 per-utterance losses.
// ---------------------------------------------------------------------------
__global__ void k3_mean(const float* __restrict__ negll, float* __restrict__ out)
{
    if (threadIdx.x == 0) {
        float s = 0.0f;
        #pragma unroll
        for (int b = 0; b < Bsz; ++b) s += negll[b];
        out[0] = s / (float)Bsz;
    }
}

// ---------------------------------------------------------------------------
extern "C" void kernel_launch(void* const* d_in, const int* in_sizes, int n_in,
                              void* d_out, int out_size, void* d_ws, size_t ws_size,
                              hipStream_t stream)
{
    const float* logits     = (const float*)d_in[0];
    const int*   y          = (const int*)  d_in[1];
    const int*   logit_lens = (const int*)  d_in[2];
    const int*   y_lens     = (const int*)  d_in[3];
    // d_in[4] = batch_offset, d_in[5] = max_f_len : packing metadata, unused.

    float* blank_ws = (float*)d_ws;                       // B*T*(U+1) floats
    float* emit_ws  = blank_ws + TOTAL_ROWS;              // B*T*U floats
    float* negll    = emit_ws + (size_t)Bsz * Tsz * Usz;  // B floats

    const int k1_blocks = TOTAL_ROWS / (16 * K1_WAVES);   // 8256, exact
    k1_softmax_gather<<<k1_blocks, 128, 0, stream>>>(logits, y, blank_ws, emit_ws);
    k2_alpha_wavefront<<<Bsz, K2_THREADS, 0, stream>>>(blank_ws, emit_ws,
                                                       logit_lens, y_lens, negll);
    k3_mean<<<1, 32, 0, stream>>>(negll, (float*)d_out);
}
